// GPSLayer_82712480187094
// MI455X (gfx1250) — matrix operand order repaired and checked by reading, not compiled, not run on hardware
//
#include <hip/hip_runtime.h>
#include <hip/hip_bf16.h>
#include <math.h>

// ---------------------------------------------------------------------------
// GraphGPS layer for MI455X (gfx1250, wave32, WMMA).
// GEMMs use v_wmma_f32_16x16x32_f16. All WMMA operands are pre-converted to
// f16 with k-contiguous layout, so each fragment is loaded with b128s:
//   A fragment = 2 x v8h  (k = g*8 .. +7, and +16)
//   B fragment = 2 x v8h  (k = g*8 .. +7, and +16)
// No fp32->fp16 conversion and no strided scalar loads in the hot loop.
// ---------------------------------------------------------------------------

typedef _Float16 half_t;
typedef __attribute__((ext_vector_type(16))) _Float16 v16h;
typedef __attribute__((ext_vector_type(8)))  _Float16 v8h;
typedef __attribute__((ext_vector_type(8)))  float    v8f;

#define NNODES 16384
#define DIMH   128
#define NEDGES 262144
#define NGRAPH 32
#define SEQLEN 512
#define NHEAD  8
#define DHEAD  16
#define BN_EPS 1e-5f

struct GemmP {
  const half_t* A; long long a_g, a_h; int lda_m;   // A: [m][k], k contiguous
  const half_t* B; long long b_g, b_h; int ldb_n;   // B: [n][k], k contiguous
  void*         C; long long c_g, c_h; int ldc_m;
  const float* bias;
  const float* resid; int ldr_m;
  int M, Nn, Hh, batches;
  float alpha;
};

// ---------------------------------------------------------------------------
// Batched WMMA GEMM: C[b] = alpha * A[b] @ B[b]^T-ish (+bias)(+resid)(relu)
// (B stored column-of-C-major: element (k,n) at B[n*ldb_n + k].)
// ---------------------------------------------------------------------------
template <int K, int NACC, bool BIAS, bool RESID, bool RELU, bool C16>
__global__ void wmma_gemm(GemmP p) {
  constexpr bool PAD    = (K % 32) != 0;   // only K=16 (dh) case
  constexpr int  KSTEPS = (K + 31) / 32;
  const int lane   = threadIdx.x;               // 0..31
  const int wv     = threadIdx.y;               // 0..1
  const int tile_m = blockIdx.x;
  const int n0     = (blockIdx.y * 2 + wv) * (16 * NACC);
  if (n0 >= p.Nn) return;                       // wave-uniform exit
  const int bb = blockIdx.z;
  const int gg = bb / p.Hh;
  const int hh = bb % p.Hh;
  const half_t* __restrict__ A = p.A + (long long)gg * p.a_g + (long long)hh * p.a_h;
  const half_t* __restrict__ B = p.B + (long long)gg * p.b_g + (long long)hh * p.b_h;
  const long long coff = (long long)gg * p.c_g + (long long)hh * p.c_h;

  const int g = lane >> 4;                      // lane group
  const int l = lane & 15;
  const long long am = (long long)(tile_m * 16 + l) * p.lda_m;
  long long bn[NACC];
  #pragma unroll
  for (int j = 0; j < NACC; ++j)
    bn[j] = (long long)(n0 + j * 16 + l) * p.ldb_n;

  v8f acc[NACC];
  #pragma unroll
  for (int j = 0; j < NACC; ++j) acc[j] = (v8f){};

  #pragma unroll
  for (int ks = 0; ks < KSTEPS; ++ks) {
    const int k0 = ks * 32;
    // ---- A fragment: two contiguous 8-half runs (b128 each) ----
    const v8h alo = *(const v8h*)(A + am + k0 + g * 8);
    v8h ahi;
    if (!PAD) ahi = *(const v8h*)(A + am + k0 + 16 + g * 8);
    else      ahi = (v8h){};                    // k >= 16 zero-padded
    const v16h va = __builtin_shufflevector(alo, ahi,
        0,1,2,3,4,5,6,7,8,9,10,11,12,13,14,15);
    // ---- B fragments + WMMA, A reused NACC times ----
    #pragma unroll
    for (int j = 0; j < NACC; ++j) {
      const v8h blo = *(const v8h*)(B + bn[j] + k0 + g * 8);
      const v8h bhi = *(const v8h*)(B + bn[j] + k0 + 16 + g * 8);
      const v16h vb = __builtin_shufflevector(blo, bhi,
          0,1,2,3,4,5,6,7,8,9,10,11,12,13,14,15);
      acc[j] = __builtin_amdgcn_wmma_f32_16x16x32_f16(
          false, va, false, vb, (short)0, acc[j], false, false);
    }
  }

  // ---- epilogue: lane l holds column n; VGPR v holds row v + 8*g ----
  #pragma unroll
  for (int j = 0; j < NACC; ++j) {
    const int n = n0 + j * 16 + l;
    const float bv = BIAS ? p.bias[n] : 0.f;
    #pragma unroll
    for (int v = 0; v < 8; ++v) {
      const int m = tile_m * 16 + v + 8 * g;
      float val = acc[j][v] * p.alpha + bv;
      if (RESID) val += p.resid[(long long)m * p.ldr_m + n];
      if (RELU)  val = fmaxf(val, 0.f);
      const long long ci = coff + (long long)m * p.ldc_m + n;
      if (C16) ((half_t*)p.C)[ci] = (half_t)val;
      else     ((float*) p.C)[ci] = val;
    }
  }
}

template <int K, int NACC, bool BIAS, bool RESID, bool RELU, bool C16>
static void launch_gemm(const GemmP& p, hipStream_t s) {
  dim3 blk(32, 2);
  const int colsPerBlock = 2 * 16 * NACC;
  dim3 grd((unsigned)((p.M + 15) / 16),
           (unsigned)((p.Nn + colsPerBlock - 1) / colsPerBlock),
           (unsigned)p.batches);
  wmma_gemm<K, NACC, BIAS, RESID, RELU, C16><<<grd, blk, 0, s>>>(p);
}

// ---------------------------------------------------------------------------
// Edge aggregation: aggr[dst] += relu(x[src] + edge_attr).  One wave per edge,
// each lane handles 4 channels (float4), scatter via global float atomics.
// ---------------------------------------------------------------------------
__global__ void edge_aggr(const float* __restrict__ x, const int* __restrict__ src,
                          const int* __restrict__ dst, const float* __restrict__ ea,
                          float* __restrict__ aggr) {
  const int e = blockIdx.x * blockDim.y + threadIdx.y;
  if (e >= NEDGES) return;
  const int c = threadIdx.x * 4;
  if (e + 8 < NEDGES)
    __builtin_prefetch(ea + (long long)(e + 8) * DIMH + c, 0, 1);
  const int s = src[e];
  const int d = dst[e];
  const float4 xs = *(const float4*)(x + (long long)s * DIMH + c);
  const float4 ev = *(const float4*)(ea + (long long)e * DIMH + c);
  float* ap = aggr + (long long)d * DIMH + c;
  atomicAdd(ap + 0, fmaxf(xs.x + ev.x, 0.f));
  atomicAdd(ap + 1, fmaxf(xs.y + ev.y, 0.f));
  atomicAdd(ap + 2, fmaxf(xs.z + ev.z, 0.f));
  atomicAdd(ap + 3, fmaxf(xs.w + ev.w, 0.f));
}

// ---------------------------------------------------------------------------
// Elementwise helpers (conversion producers for the f16 GEMM operands)
// ---------------------------------------------------------------------------
__global__ void cvt16(const float* __restrict__ in, half_t* __restrict__ out, int total) {
  const int i = blockIdx.x * blockDim.x + threadIdx.x;
  if (i < total) out[i] = (half_t)in[i];
}

// weight [K][N] row-major -> f16 [N][K] (k contiguous)
__global__ void wt_t16(const float* __restrict__ in, half_t* __restrict__ out,
                       int K, int N) {
  const int o = blockIdx.x * blockDim.x + threadIdx.x;
  if (o >= K * N) return;
  const int n = o / K, k = o % K;
  out[o] = (half_t)in[k * N + n];
}

// z16 = f16(x + aggr)
__global__ void add_cvt16(const float* __restrict__ a, const float* __restrict__ b,
                          half_t* __restrict__ out, int total) {
  const int i = blockIdx.x * blockDim.x + threadIdx.x;
  if (i < total) out[i] = (half_t)(a[i] + b[i]);
}

// hsum = a+b (f32) and hsum16 = f16(a+b)
__global__ void add_both(const float* __restrict__ a, const float* __restrict__ b,
                         float* __restrict__ o32, half_t* __restrict__ o16, int total) {
  const int i = blockIdx.x * blockDim.x + threadIdx.x;
  if (i < total) { const float s = a[i] + b[i]; o32[i] = s; o16[i] = (half_t)s; }
}

// Vt16[((g*H+h)*16 + d)*512 + kv] = qkv16[(g*512+kv)*384 + 256 + h*16 + d]
__global__ void v_transpose(const half_t* __restrict__ qkv16, half_t* __restrict__ vt) {
  const int o = blockIdx.x * blockDim.x + threadIdx.x;
  if (o >= NGRAPH * NHEAD * DHEAD * SEQLEN) return;
  const int kv = o & (SEQLEN - 1);
  const int d  = (o >> 9) & (DHEAD - 1);
  const int bb = o >> 13;
  const int h  = bb & (NHEAD - 1);
  const int g  = bb >> 3;
  vt[o] = qkv16[((long long)(g * SEQLEN + kv)) * (3 * DIMH) + 2 * DIMH + h * DHEAD + d];
}

// ---------------------------------------------------------------------------
// BatchNorm: column stats (sum, sumsq) then normalize. 128 columns.
// ---------------------------------------------------------------------------
__global__ void bn_stats(const float* __restrict__ x, float* __restrict__ stats) {
  const int col = threadIdx.x;                // blockDim.x == 128
  const int r0  = blockIdx.x * 128;
  float s = 0.f, s2 = 0.f;
  for (int r = r0; r < r0 + 128; ++r) {
    const float v = x[(long long)r * DIMH + col];
    s += v; s2 += v * v;
  }
  atomicAdd(&stats[col], s);
  atomicAdd(&stats[DIMH + col], s2);
}

__global__ void bn_apply(const float* __restrict__ x, const float* __restrict__ stats,
                         const float* __restrict__ gm, const float* __restrict__ bt,
                         float* __restrict__ out, int total) {
  const int i = blockIdx.x * blockDim.x + threadIdx.x;
  if (i >= total) return;
  const int col = i & (DIMH - 1);
  const float inv_n = 1.f / (float)NNODES;
  const float mu  = stats[col] * inv_n;
  const float var = stats[DIMH + col] * inv_n - mu * mu;   // biased, as torch
  out[i] = (x[i] - mu) * rsqrtf(var + BN_EPS) * gm[col] + bt[col];
}

// ---------------------------------------------------------------------------
// Softmax over rows of 512 (f32 in, f16 out). One wave per row, 16 elems/lane,
// wave32 shuffle reductions.
// ---------------------------------------------------------------------------
__global__ void softmax512_f16(const float* __restrict__ s, half_t* __restrict__ o,
                               int rows) {
  const int row = blockIdx.x * blockDim.y + threadIdx.y;
  if (row >= rows) return;
  const float* p = s + (long long)row * 512;
  half_t*      q = o + (long long)row * 512;
  const int lane = threadIdx.x;
  float v[16];
  float mx = -3.4e38f;
  #pragma unroll
  for (int j = 0; j < 16; ++j) { v[j] = p[lane + 32 * j]; mx = fmaxf(mx, v[j]); }
  #pragma unroll
  for (int off = 16; off > 0; off >>= 1) mx = fmaxf(mx, __shfl_xor(mx, off, 32));
  float sum = 0.f;
  #pragma unroll
  for (int j = 0; j < 16; ++j) { v[j] = __expf(v[j] - mx); sum += v[j]; }
  #pragma unroll
  for (int off = 16; off > 0; off >>= 1) sum += __shfl_xor(sum, off, 32);
  const float inv = 1.f / sum;
  #pragma unroll
  for (int j = 0; j < 16; ++j) q[lane + 32 * j] = (half_t)(v[j] * inv);
}

// ---------------------------------------------------------------------------
// Host-side launch composition
// ---------------------------------------------------------------------------
static GemmP simple_gemm(const half_t* A, int lda, const half_t* B, int ldbn,
                         void* C, int ldc, const float* bias, const float* resid,
                         int M, int Nn) {
  GemmP p{};
  p.A = A; p.a_g = 0; p.a_h = 0; p.lda_m = lda;
  p.B = B; p.b_g = 0; p.b_h = 0; p.ldb_n = ldbn;
  p.C = C; p.c_g = 0; p.c_h = 0; p.ldc_m = ldc;
  p.bias = bias; p.resid = resid; p.ldr_m = ldc;
  p.M = M; p.Nn = Nn; p.Hh = 1; p.batches = 1;
  p.alpha = 1.f;
  return p;
}

extern "C" void kernel_launch(void* const* d_in, const int* in_sizes, int n_in,
                              void* d_out, int out_size, void* d_ws, size_t ws_size,
                              hipStream_t stream) {
  (void)in_sizes; (void)n_in; (void)out_size; (void)ws_size;

  const float* x          = (const float*)d_in[0];
  const int*   eidx       = (const int*)  d_in[1];
  const float* eattr      = (const float*)d_in[2];
  const float* gin_w1     = (const float*)d_in[3];
  const float* gin_b1     = (const float*)d_in[4];
  const float* gin_w2     = (const float*)d_in[5];
  const float* gin_b2     = (const float*)d_in[6];
  const float* bn1l_g     = (const float*)d_in[7];
  const float* bn1l_b     = (const float*)d_in[8];
  const float* in_proj_w  = (const float*)d_in[9];
  const float* in_proj_b  = (const float*)d_in[10];
  const float* out_proj_w = (const float*)d_in[11];
  const float* out_proj_b = (const float*)d_in[12];
  const float* bn1a_g     = (const float*)d_in[13];
  const float* bn1a_b     = (const float*)d_in[14];
  const float* ff_w1      = (const float*)d_in[15];
  const float* ff_b1      = (const float*)d_in[16];
  const float* ff_w2      = (const float*)d_in[17];
  const float* ff_b2      = (const float*)d_in[18];
  const float* bn2_g      = (const float*)d_in[19];
  const float* bn2_b      = (const float*)d_in[20];
  float* out = (float*)d_out;

  const int* esrc = eidx;
  const int* edst = eidx + NEDGES;

  // ---- workspace carve-up (bytes, 256B aligned) ----
  char*  Wb  = (char*)d_ws;
  size_t off = 0;
  auto carve = [&](size_t bytes) {
    void* p = Wb + off;
    off += (bytes + 255) & ~(size_t)255;
    return p;
  };
  const size_t ND = (size_t)NNODES * DIMH;
  float*  aggr    = (float*) carve(ND * 4);
  float*  hlocal  = (float*) carve(ND * 4);
  float*  hattn   = (float*) carve(ND * 4);
  float*  hsum    = (float*) carve(ND * 4);
  float*  h2      = (float*) carve(ND * 4);
  float*  scores  = (float*) carve((size_t)NGRAPH * NHEAD * SEQLEN * SEQLEN * 4);
  float*  stats   = (float*) carve(2 * DIMH * 4);
  half_t* x16     = (half_t*)carve(ND * 2);
  half_t* z16     = (half_t*)carve(ND * 2);
  half_t* t1_16   = (half_t*)carve(ND * 2);
  half_t* qkv16   = (half_t*)carve((size_t)NNODES * 3 * DIMH * 2);
  half_t* attn16  = (half_t*)carve((size_t)NGRAPH * NHEAD * SEQLEN * SEQLEN * 2);
  half_t* vt16    = (half_t*)carve((size_t)NGRAPH * NHEAD * DHEAD * SEQLEN * 2);
  half_t* o16     = (half_t*)carve(ND * 2);
  half_t* hsum16  = (half_t*)carve(ND * 2);
  half_t* t2_16   = (half_t*)carve((size_t)NNODES * 2 * DIMH * 2);
  half_t* gw1t    = (half_t*)carve((size_t)DIMH * DIMH * 2);
  half_t* gw2t    = (half_t*)carve((size_t)DIMH * DIMH * 2);
  half_t* ipw16   = (half_t*)carve((size_t)3 * DIMH * DIMH * 2);
  half_t* opw16   = (half_t*)carve((size_t)DIMH * DIMH * 2);
  half_t* fw1t    = (half_t*)carve((size_t)DIMH * 2 * DIMH * 2);
  half_t* fw2t    = (half_t*)carve((size_t)2 * DIMH * DIMH * 2);

  const int totalND = (int)ND;
  #define GRID1D(n) dim3((unsigned)(((n) + 255) / 256)), dim3(256)

  // ---- 0. one-time operand preparation (f16, k-contiguous layouts) ----
  cvt16 <<<GRID1D(totalND), 0, stream>>>(x, x16, totalND);
  wt_t16<<<GRID1D(DIMH * DIMH), 0, stream>>>(gin_w1, gw1t, DIMH, DIMH);
  wt_t16<<<GRID1D(DIMH * DIMH), 0, stream>>>(gin_w2, gw2t, DIMH, DIMH);
  cvt16 <<<GRID1D(3 * DIMH * DIMH), 0, stream>>>(in_proj_w, ipw16, 3 * DIMH * DIMH);
  cvt16 <<<GRID1D(DIMH * DIMH), 0, stream>>>(out_proj_w, opw16, DIMH * DIMH);
  wt_t16<<<GRID1D(DIMH * 2 * DIMH), 0, stream>>>(ff_w1, fw1t, DIMH, 2 * DIMH);
  wt_t16<<<GRID1D(2 * DIMH * DIMH), 0, stream>>>(ff_w2, fw2t, 2 * DIMH, DIMH);

  // ---- 1. GINE aggregation: aggr = segment_sum(relu(x[src]+e), dst) ----
  hipMemsetAsync(aggr, 0, ND * sizeof(float), stream);
  {
    dim3 eblk(32, 8);
    edge_aggr<<<NEDGES / 8, eblk, 0, stream>>>(x, esrc, edst, eattr, aggr);
  }
  // ---- 2. z16 = f16(x + aggr) ----
  add_cvt16<<<GRID1D(totalND), 0, stream>>>(x, aggr, z16, totalND);

  // ---- 3. t1 = relu(z @ gin_w1 + b1)  -> f16 ----
  launch_gemm<DIMH, 4, true, false, true, true>(
      simple_gemm(z16, DIMH, gw1t, DIMH, t1_16, DIMH, gin_b1, nullptr,
                  NNODES, DIMH), stream);
  // ---- 4. h_local_pre = t1 @ gin_w2 + b2 + x  -> f32 ----
  launch_gemm<DIMH, 4, true, true, false, false>(
      simple_gemm(t1_16, DIMH, gw2t, DIMH, hlocal, DIMH, gin_b2, x,
                  NNODES, DIMH), stream);
  // ---- 5. BN(h_local) ----
  hipMemsetAsync(stats, 0, 2 * DIMH * sizeof(float), stream);
  bn_stats<<<NNODES / 128, DIMH, 0, stream>>>(hlocal, stats);
  bn_apply<<<GRID1D(totalND), 0, stream>>>(hlocal, stats, bn1l_g, bn1l_b, hlocal, totalND);

  // ---- 6. qkv = x @ in_proj_w^T + in_proj_b  -> f16 ----
  launch_gemm<DIMH, 4, true, false, false, true>(
      simple_gemm(x16, DIMH, ipw16, DIMH, qkv16, 3 * DIMH, in_proj_b, nullptr,
                  NNODES, 3 * DIMH), stream);

  // ---- 7. scores[g,h] = Q K^T / sqrt(dh)  (K=16 padded to 32) -> f32 ----
  {
    GemmP p{};
    p.A = qkv16;        p.a_g = (long long)SEQLEN * 3 * DIMH; p.a_h = DHEAD;
    p.lda_m = 3 * DIMH;
    p.B = qkv16 + DIMH; p.b_g = (long long)SEQLEN * 3 * DIMH; p.b_h = DHEAD;
    p.ldb_n = 3 * DIMH;                                  // implicit K^T
    p.C = scores;       p.c_g = (long long)NHEAD * SEQLEN * SEQLEN;
    p.c_h = (long long)SEQLEN * SEQLEN;
    p.ldc_m = SEQLEN;
    p.bias = nullptr; p.resid = nullptr; p.ldr_m = 0;
    p.M = SEQLEN; p.Nn = SEQLEN; p.Hh = NHEAD; p.batches = NGRAPH * NHEAD;
    p.alpha = 0.25f;  // 1/sqrt(16)
    launch_gemm<DHEAD, 4, false, false, false, false>(p, stream);
  }

  // ---- 8. softmax over k -> f16 attention weights ----
  {
    const int rows = NGRAPH * NHEAD * SEQLEN;
    dim3 sblk(32, 8);
    softmax512_f16<<<rows / 8, sblk, 0, stream>>>(scores, attn16, rows);
  }

  // ---- 8b. V^T (f16, [g,h][d][kv]) for k-contiguous attn@V ----
  v_transpose<<<GRID1D(NGRAPH * NHEAD * DHEAD * SEQLEN), 0, stream>>>(qkv16, vt16);

  // ---- 9. o[g,h] = attn @ V  (K=512, N=16) -> f16 ----
  {
    GemmP p{};
    p.A = attn16;  p.a_g = (long long)NHEAD * SEQLEN * SEQLEN;
    p.a_h = (long long)SEQLEN * SEQLEN;
    p.lda_m = SEQLEN;
    p.B = vt16;    p.b_g = (long long)NHEAD * DHEAD * SEQLEN;
    p.b_h = (long long)DHEAD * SEQLEN;
    p.ldb_n = SEQLEN;
    p.C = o16;     p.c_g = (long long)SEQLEN * DIMH; p.c_h = DHEAD;
    p.ldc_m = DIMH;
    p.bias = nullptr; p.resid = nullptr; p.ldr_m = 0;
    p.M = SEQLEN; p.Nn = DHEAD; p.Hh = NHEAD; p.batches = NGRAPH * NHEAD;
    p.alpha = 1.f;
    launch_gemm<SEQLEN, 1, false, false, false, true>(p, stream);
  }

  // ---- 10. h_attn_pre = o @ out_proj_w^T + out_proj_b + x  -> f32 ----
  launch_gemm<DIMH, 4, true, true, false, false>(
      simple_gemm(o16, DIMH, opw16, DIMH, hattn, DIMH, out_proj_b, x,
                  NNODES, DIMH), stream);
  // ---- 11. BN(h_attn) ----
  hipMemsetAsync(stats, 0, 2 * DIMH * sizeof(float), stream);
  bn_stats<<<NNODES / 128, DIMH, 0, stream>>>(hattn, stats);
  bn_apply<<<GRID1D(totalND), 0, stream>>>(hattn, stats, bn1a_g, bn1a_b, hattn, totalND);

  // ---- 12. h = h_local + h_attn  (f32 + f16 copies) ----
  add_both<<<GRID1D(totalND), 0, stream>>>(hlocal, hattn, hsum, hsum16, totalND);

  // ---- 13. t2 = relu(h @ ff_w1 + ff_b1)  -> f16  [N,128]x[128,256] ----
  launch_gemm<DIMH, 4, true, false, true, true>(
      simple_gemm(hsum16, DIMH, fw1t, DIMH, t2_16, 2 * DIMH, ff_b1, nullptr,
                  NNODES, 2 * DIMH), stream);
  // ---- 14. h2 = t2 @ ff_w2 + ff_b2 + h  -> f32  [N,256]x[256,128] ----
  launch_gemm<2 * DIMH, 4, true, true, false, false>(
      simple_gemm(t2_16, 2 * DIMH, fw2t, 2 * DIMH, h2, DIMH, ff_b2, hsum,
                  NNODES, DIMH), stream);
  // ---- 15. BN -> out ----
  hipMemsetAsync(stats, 0, 2 * DIMH * sizeof(float), stream);
  bn_stats<<<NNODES / 128, DIMH, 0, stream>>>(h2, stats);
  bn_apply<<<GRID1D(totalND), 0, stream>>>(h2, stats, bn2_g, bn2_b, out, totalND);
  #undef GRID1D
}
